// VLMDecoderLayer_23613730194151
// MI455X (gfx1250) — compile-verified, matrix operand order checked by reading, not run on hardware
//
#include <hip/hip_runtime.h>
#include <hip/hip_bf16.h>

#define SEQ   2048
#define HIDN  4096
#define NH    32
#define NKV   8
#define HD    128
#define INTER 14336

typedef __attribute__((ext_vector_type(16))) __bf16 v16bf;
typedef __attribute__((ext_vector_type(8)))  float  v8f;
typedef __attribute__((ext_vector_type(4)))  unsigned int u32x4;
typedef __attribute__((ext_vector_type(4)))  int          i32x4;
typedef __attribute__((ext_vector_type(8)))  int          i32x8;

#if __has_builtin(__builtin_amdgcn_tensor_load_to_lds) && __has_builtin(__builtin_amdgcn_s_wait_tensorcnt)
#define HAVE_TDM 1
#else
#define HAVE_TDM 0
#endif

union FragU  { unsigned int u[8]; v16bf v; };
union U4E8   { uint4 q; unsigned short e[8]; };

__device__ __forceinline__ unsigned short f2bf(float f) {
  unsigned int u = __builtin_bit_cast(unsigned int, f);
  u += 0x7FFFu + ((u >> 16) & 1u);  // round-to-nearest-even
  return (unsigned short)(u >> 16);
}
__device__ __forceinline__ float bf2f(unsigned short h) {
  return __builtin_bit_cast(float, ((unsigned int)h) << 16);
}
__device__ __forceinline__ unsigned int pack2(float a, float b) {
  return (unsigned int)f2bf(a) | ((unsigned int)f2bf(b) << 16);
}

#if HAVE_TDM
// Issue a 2D TDM load: tile (tile_w x tile_h) of bf16 from a row-major tensor
// with row stride `stride` (elements) into LDS at byte offset lds_off.
// D# per CDNA5 ISA 8.3-8.6. Must be called by exactly one wave (per-wave DMA,
// EXEC ignored); args must be wave-uniform. 6-arg builtin (clang-23 flavor):
// (g0, g1, g2, g3, g4, cpol) with trailing groups zero => 2D tensor.
__device__ __forceinline__ void tdm_load_2d(unsigned lds_off, const void* gaddr,
                                            unsigned tile_w, unsigned tile_h,
                                            unsigned stride,
                                            unsigned tdim0, unsigned tdim1) {
  unsigned long long ga = (unsigned long long)gaddr;
  u32x4 g0;
  g0[0] = 1u;                                    // count=1, user descriptor
  g0[1] = lds_off;                               // lds_addr (bytes)
  g0[2] = (unsigned)ga;                          // global_addr[31:0]
  g0[3] = (unsigned)(ga >> 32) | (2u << 30);     // global_addr[56:32] | type=2
  i32x8 g1;
  g1[0] = (int)(1u << 16);                       // wg_mask=0, data_size=1 (2B)
  g1[1] = (int)((tdim0 & 0xFFFFu) << 16);        // tensor_dim0[15:0]
  g1[2] = (int)((tdim0 >> 16) | ((tdim1 & 0xFFFFu) << 16));  // dim0 hi | dim1 lo
  g1[3] = (int)((tdim1 >> 16) | (tile_w << 16)); // dim1 hi | tile_dim0
  g1[4] = (int)(tile_h & 0xFFFFu);               // tile_dim1, tile_dim2=0
  g1[5] = (int)stride;                           // tensor_dim0_stride[31:0]
  g1[6] = 0;
  g1[7] = 0;
  i32x4 z4 = {0, 0, 0, 0};
  i32x8 z8 = {0, 0, 0, 0, 0, 0, 0, 0};
  __builtin_amdgcn_tensor_load_to_lds(g0, g1, z4, z4, z8, 0);
}
#endif

// 16x32 bf16 WMMA fragment from an LDS [row][k] tile (ISA 7.12.2 layout).
__device__ __forceinline__ v16bf ldfrag(const unsigned short* lds, int rowBase,
                                        int k0, int strideHalf) {
  const int lane = threadIdx.x & 31;
  const unsigned int* p =
      (const unsigned int*)(lds + (size_t)(rowBase + (lane & 15)) * strideHalf) + (k0 >> 1);
  const int h4 = (lane >> 4) << 2;
  FragU f;
#pragma unroll
  for (int q = 0; q < 8; ++q) {
    int kd = (q & 3) + h4 + ((q >> 2) << 3);
    f.u[q] = p[kd];
  }
  return f.v;
}

// ---------------------------------------------------------------------------
// GEMM: C[M,N] = A[M,K]*B[K,N] (+addsrc). A,B bf16; C fp32 or bf16.
// 128x128x32 tile / 256 threads (8 waves). A tile staged via TDM.
// ---------------------------------------------------------------------------
template <bool BFOUT>
__global__ __launch_bounds__(256) void gemm_bf16(
    const unsigned short* __restrict__ A, const unsigned short* __restrict__ B,
    const float* __restrict__ addsrc, float* __restrict__ Cf,
    unsigned short* __restrict__ Cb, int M, int N, int K) {
  __shared__ unsigned short Ab[128 * 32];   // [m][k]
  __shared__ unsigned short Bb[128 * 32];   // [n][k] (transposed)

  const int tid = threadIdx.x;
  const int lane = tid & 31;
  const int n0 = blockIdx.x * 128;
  const int m0 = blockIdx.y * 128;
  const int w  = tid >> 5;
  const int wm = (w >> 2) * 64;
  const int wn = (w & 3) * 32;

  v8f acc[4][2];
#pragma unroll
  for (int i = 0; i < 4; ++i)
#pragma unroll
    for (int j = 0; j < 2; ++j)
#pragma unroll
      for (int e = 0; e < 8; ++e) acc[i][j][e] = 0.f;

#if HAVE_TDM
  const unsigned ldsA = (unsigned)(unsigned long long)(const void*)Ab;
#endif

  for (int k0 = 0; k0 < K; k0 += 32) {
#if HAVE_TDM
    if (tid < 32)  // wave 0 drives the DMA for the A tile
      tdm_load_2d(ldsA, A + (size_t)m0 * K + k0, 32, 128, (unsigned)K,
                  (unsigned)K, (unsigned)M);
#else
#pragma unroll
    for (int t = 0; t < 2; ++t) {
      int i = tid + t * 256;            // uint4 id (8 bf16), 512 total
      int r = i >> 2;                   // 4 uint4 per 32-el row
      int c = (i & 3) << 3;
      *(uint4*)(Ab + r * 32 + c) = *(const uint4*)(A + (size_t)(m0 + r) * K + k0 + c);
    }
#endif
    if (k0 + 32 < K && tid < 32)        // prefetch next B tile rows
      __builtin_prefetch(B + (size_t)(k0 + 32 + tid) * N + n0, 0, 0);
    // stage B transposed: 32x128 bf16 -> Bb[n][k]
#pragma unroll
    for (int t = 0; t < 2; ++t) {
      int i = tid + t * 256;            // uint4 id, 512 total
      int r = i >> 4;                   // k row 0..31 (16 uint4 per 128-el row)
      int c = (i & 15) << 3;            // n
      U4E8 f;
      f.q = *(const uint4*)(B + (size_t)(k0 + r) * N + n0 + c);
#pragma unroll
      for (int j = 0; j < 8; ++j) Bb[(c + j) * 32 + r] = f.e[j];
    }
#if HAVE_TDM
    if (tid < 32) __builtin_amdgcn_s_wait_tensorcnt(0);
#endif
    __syncthreads();

    v16bf bq[2];
    bq[0] = ldfrag(Bb, wn + 0,  0, 32);
    bq[1] = ldfrag(Bb, wn + 16, 0, 32);
#pragma unroll
    for (int i = 0; i < 4; ++i) {
      v16bf aq = ldfrag(Ab, wm + i * 16, 0, 32);
#pragma unroll
      for (int j = 0; j < 2; ++j)
        acc[i][j] = __builtin_amdgcn_wmma_f32_16x16x32_bf16(
            false, aq, false, bq[j], (short)0, acc[i][j], false, false);
    }
    __syncthreads();
  }

  const int nn = n0 + wn + (lane & 15);
  const int mb = m0 + wm + 8 * (lane >> 4);
#pragma unroll
  for (int i = 0; i < 4; ++i)
#pragma unroll
    for (int j = 0; j < 2; ++j)
#pragma unroll
      for (int r = 0; r < 8; ++r) {
        size_t idx = (size_t)(mb + i * 16 + r) * N + nn + j * 16;
        float vv = acc[i][j][r];
        if (addsrc) vv += addsrc[idx];
        if (BFOUT) Cb[idx] = f2bf(vv);
        else       Cf[idx] = vv;
      }
}

// ---------------------------------------------------------------------------
// Flash attention (causal, GQA 4:1): Q,K,V bf16 in; attn out bf16.
// Q/K tiles staged via TDM; V staged manually (needs transpose).
// Block: 128 threads = 4 waves; each wave owns 16 query rows.
// ---------------------------------------------------------------------------
__global__ __launch_bounds__(128) void flash_attn(
    const unsigned short* __restrict__ Q, const unsigned short* __restrict__ Kb,
    const unsigned short* __restrict__ Vb, unsigned short* __restrict__ O) {
  __shared__ unsigned short Qs[64 * 128];    // [q][d]
  __shared__ unsigned short Ks[32 * 128];    // [key][d]
  __shared__ unsigned short Vst[128 * 32];   // [d][key]
  __shared__ unsigned short Ps[4][16 * 32];  // per-wave P tile

  const int tid  = threadIdx.x;
  const int lane = tid & 31;
  const int w    = tid >> 5;
  const int q0   = blockIdx.x * 64;
  const int h    = blockIdx.y;
  const int kvh  = h >> 2;
  const float scl = 0.08838834764831845f;    // 1/sqrt(128), folded into scores

#if HAVE_TDM
  const unsigned ldsQ = (unsigned)(unsigned long long)(const void*)Qs;
  const unsigned ldsK = (unsigned)(unsigned long long)(const void*)Ks;
  if (tid < 32)
    tdm_load_2d(ldsQ, Q + (size_t)q0 * (NH * HD) + h * HD, 128, 64, NH * HD,
                NH * HD, SEQ);
#else
#pragma unroll
  for (int t = 0; t < 8; ++t) {
    int i = tid + t * 128;                   // uint4 id, 1024
    int r = i >> 4;
    int c = (i & 15) << 3;
    *(uint4*)(Qs + r * 128 + c) = *(const uint4*)(Q + (size_t)(q0 + r) * (NH * HD) + h * HD + c);
  }
#endif

  float Mr[8], Lr[8];
  v8f o[8];
#pragma unroll
  for (int r = 0; r < 8; ++r) { Mr[r] = -3.0e38f; Lr[r] = 0.f; }
#pragma unroll
  for (int d = 0; d < 8; ++d)
#pragma unroll
    for (int e = 0; e < 8; ++e) o[d][e] = 0.f;

  const int qrow = q0 + w * 16;

  for (int k0 = 0; k0 < q0 + 64; k0 += 32) {
    __syncthreads();  // previous iteration done with Ks/Vst (and Qs staged)
#if HAVE_TDM
    if (tid < 32)
      tdm_load_2d(ldsK, Kb + (size_t)k0 * (NKV * HD) + kvh * HD, 128, 32,
                  NKV * HD, NKV * HD, SEQ);
#else
#pragma unroll
    for (int t = 0; t < 4; ++t) {
      int i = tid + t * 128;                 // uint4 id, 512
      int r = i >> 4;
      int c = (i & 15) << 3;
      *(uint4*)(Ks + r * 128 + c) = *(const uint4*)(Kb + (size_t)(k0 + r) * (NKV * HD) + kvh * HD + c);
    }
#endif
    // V transposed (manual scatter)
#pragma unroll
    for (int t = 0; t < 4; ++t) {
      int i = tid + t * 128;                 // uint4 id, 512
      int r = i >> 4;                        // key 0..31
      int c = (i & 15) << 3;                 // d
      U4E8 f;
      f.q = *(const uint4*)(Vb + (size_t)(k0 + r) * (NKV * HD) + kvh * HD + c);
#pragma unroll
      for (int j = 0; j < 8; ++j) Vst[(c + j) * 32 + r] = f.e[j];
    }
#if HAVE_TDM
    if (tid < 32) __builtin_amdgcn_s_wait_tensorcnt(0);
#endif
    __syncthreads();

    // scores S = (Q*K^T) * scl : 16x32 per wave
    v8f s[2];
#pragma unroll
    for (int j = 0; j < 2; ++j)
#pragma unroll
      for (int e = 0; e < 8; ++e) s[j][e] = 0.f;
#pragma unroll
    for (int kc = 0; kc < 4; ++kc) {
      v16bf qa = ldfrag(Qs, w * 16, kc * 32, 128);
#pragma unroll
      for (int j = 0; j < 2; ++j) {
        v16bf kb = ldfrag(Ks, j * 16, kc * 32, 128);
        s[j] = __builtin_amdgcn_wmma_f32_16x16x32_bf16(
            false, qa, false, kb, (short)0, s[j], false, false);
      }
    }
    // scale + causal mask
    const int qbase = qrow + 8 * (lane >> 4);
#pragma unroll
    for (int j = 0; j < 2; ++j) {
      int key = k0 + j * 16 + (lane & 15);
#pragma unroll
      for (int r = 0; r < 8; ++r) {
        s[j][r] *= scl;
        if (key > qbase + r) s[j][r] = -3.0e38f;
      }
    }
    // online softmax across the 16-lane column groups
#pragma unroll
    for (int r = 0; r < 8; ++r) {
      float mx = fmaxf(s[0][r], s[1][r]);
#pragma unroll
      for (int off = 8; off; off >>= 1) mx = fmaxf(mx, __shfl_xor(mx, off, 32));
      float newM = fmaxf(Mr[r], mx);
      float corr = __expf(Mr[r] - newM);
      float p0 = __expf(s[0][r] - newM);
      float p1 = __expf(s[1][r] - newM);
      s[0][r] = p0; s[1][r] = p1;
      float l = p0 + p1;
#pragma unroll
      for (int off = 8; off; off >>= 1) l += __shfl_xor(l, off, 32);
      Lr[r] = Lr[r] * corr + l;
      Mr[r] = newM;
#pragma unroll
      for (int d = 0; d < 8; ++d) o[d][r] *= corr;
    }
    // P: C-layout -> A-layout via per-wave LDS bounce
    {
      unsigned short* pp = Ps[w];
      int col  = lane & 15;
      int rowh = 8 * (lane >> 4);
#pragma unroll
      for (int j = 0; j < 2; ++j)
#pragma unroll
        for (int r = 0; r < 8; ++r)
          pp[(rowh + r) * 32 + j * 16 + col] = f2bf(s[j][r]);
    }
    __syncthreads();

    // O += P * V
    v16bf pa = ldfrag(Ps[w], 0, 0, 32);
#pragma unroll
    for (int d = 0; d < 8; ++d) {
      v16bf vb = ldfrag(Vst, d * 16, 0, 32);
      o[d] = __builtin_amdgcn_wmma_f32_16x16x32_bf16(
          false, pa, false, vb, (short)0, o[d], false, false);
    }
  }

  const int col  = lane & 15;
  const int rowh = qrow + 8 * (lane >> 4);
#pragma unroll
  for (int r = 0; r < 8; ++r) {
    float inv = 1.f / Lr[r];
#pragma unroll
    for (int d = 0; d < 8; ++d)
      O[(size_t)(rowh + r) * (NH * HD) + h * HD + d * 16 + col] = f2bf(o[d][r] * inv);
  }
}

// ---------------------------------------------------------------------------
// RoPE on bf16 q (32 heads) / k (8 heads) in place; emit fp32 k,v outputs.
// ---------------------------------------------------------------------------
__global__ __launch_bounds__(64) void rope_kv(
    unsigned short* __restrict__ q, unsigned short* __restrict__ k,
    const unsigned short* __restrict__ v, const float* __restrict__ cosb,
    const float* __restrict__ sinb, float* __restrict__ kout,
    float* __restrict__ vout) {
  const int s = blockIdx.x;
  const int d = threadIdx.x;  // 0..63
  const float c0 = cosb[s * HD + d], c1 = cosb[s * HD + d + 64];
  const float s0 = sinb[s * HD + d], s1 = sinb[s * HD + d + 64];
#pragma unroll 4
  for (int h = 0; h < NH; ++h) {
    size_t b = (size_t)s * (NH * HD) + h * HD;
    float x0 = bf2f(q[b + d]), x1 = bf2f(q[b + d + 64]);
    q[b + d]      = f2bf(x0 * c0 - x1 * s0);
    q[b + d + 64] = f2bf(x1 * c1 + x0 * s1);
  }
#pragma unroll
  for (int h = 0; h < NKV; ++h) {
    size_t b = (size_t)s * (NKV * HD) + h * HD;
    float x0 = bf2f(k[b + d]), x1 = bf2f(k[b + d + 64]);
    float r0 = x0 * c0 - x1 * s0;
    float r1 = x1 * c1 + x0 * s1;
    k[b + d] = f2bf(r0);   k[b + d + 64] = f2bf(r1);
    kout[b + d] = r0;      kout[b + d + 64] = r1;
    vout[b + d] = bf2f(v[b + d]);
    vout[b + d + 64] = bf2f(v[b + d + 64]);
  }
}

// ---------------------------------------------------------------------------
__global__ __launch_bounds__(256) void rmsnorm_k(
    const float* __restrict__ x, const float* __restrict__ w,
    unsigned short* __restrict__ y) {
  __shared__ float red[8];
  const int s = blockIdx.x, tid = threadIdx.x;
  const float* xr = x + (size_t)s * HIDN;
  float ss = 0.f;
#pragma unroll
  for (int i = 0; i < HIDN / 256; ++i) { float v = xr[tid + i * 256]; ss += v * v; }
#pragma unroll
  for (int off = 16; off; off >>= 1) ss += __shfl_xor(ss, off, 32);
  if ((tid & 31) == 0) red[tid >> 5] = ss;
  __syncthreads();
  float tot = (tid < 8) ? red[tid] : 0.f;
  if (tid < 32) {
#pragma unroll
    for (int off = 4; off; off >>= 1) tot += __shfl_xor(tot, off, 32);
    if (tid == 0) red[0] = tot;
  }
  __syncthreads();
  const float inv = rsqrtf(red[0] * (1.0f / HIDN) + 1e-5f);
  unsigned short* yr = y + (size_t)s * HIDN;
#pragma unroll
  for (int i = 0; i < HIDN / 256; ++i) {
    int idx = tid + i * 256;
    yr[idx] = f2bf(xr[idx] * inv * w[idx]);
  }
}

__global__ void cvt_bf16(const float* __restrict__ s, unsigned short* __restrict__ d, int n) {
  int i = (blockIdx.x * 256 + threadIdx.x) * 4;
  if (i < n) {
    float4 f = *(const float4*)(s + i);
    *(uint2*)(d + i) = make_uint2(pack2(f.x, f.y), pack2(f.z, f.w));
  }
}

__global__ void silu_mul(const unsigned short* __restrict__ g,
                         const unsigned short* __restrict__ u,
                         unsigned short* __restrict__ hs, int n) {
  int i = blockIdx.x * blockDim.x + threadIdx.x;
  if (i < n) {
    float x = bf2f(g[i]);
    hs[i] = f2bf((x / (1.f + __expf(-x))) * bf2f(u[i]));
  }
}

// ---------------------------------------------------------------------------
extern "C" void kernel_launch(void* const* d_in, const int* in_sizes, int n_in,
                              void* d_out, int out_size, void* d_ws, size_t ws_size,
                              hipStream_t stream) {
  (void)in_sizes; (void)n_in; (void)out_size; (void)ws_size;
  const float* hidden = (const float*)d_in[0];
  const float* cosb   = (const float*)d_in[1];
  const float* sinb   = (const float*)d_in[2];
  // d_in[3] = causal mask — applied analytically
  const float* ln1 = (const float*)d_in[4];
  const float* ln2 = (const float*)d_in[5];
  const float* w_q = (const float*)d_in[6];
  const float* w_k = (const float*)d_in[7];
  const float* w_v = (const float*)d_in[8];
  const float* w_o = (const float*)d_in[9];
  const float* w_g = (const float*)d_in[10];
  const float* w_u = (const float*)d_in[11];
  const float* w_d = (const float*)d_in[12];

  const size_t SH  = (size_t)SEQ * HIDN;
  const size_t SKV = (size_t)SEQ * NKV * HD;
  const size_t SI  = (size_t)SEQ * INTER;
  const size_t WQ = (size_t)HIDN * (NH * HD);
  const size_t WK = (size_t)HIDN * (NKV * HD);
  const size_t WG = (size_t)HIDN * INTER;

  // bf16 scratch layout
  unsigned short* p = (unsigned short*)d_ws;
  unsigned short* wq_b = p;  p += WQ;
  unsigned short* wk_b = p;  p += WK;
  unsigned short* wv_b = p;  p += WK;
  unsigned short* wo_b = p;  p += WQ;
  unsigned short* wg_b = p;  p += WG;
  unsigned short* wu_b = p;  p += WG;
  unsigned short* wd_b = p;  p += WG;
  unsigned short* xn   = p;  p += SH;
  unsigned short* qb   = p;  p += SH;
  unsigned short* kb   = p;  p += SKV;
  unsigned short* vb   = p;  p += SKV;
  unsigned short* attn = p;  p += SH;
  unsigned short* gate = p;  p += SI;
  unsigned short* up   = p;  p += SI;
  unsigned short* hs   = p;  p += SI;
  float* x2 = (float*)p;  // fp32 residual-2, after all bf16 buffers

  float* out_x = (float*)d_out;
  float* out_k = out_x + SH;
  float* out_v = out_k + SKV;

  // one-time (per launch) weight quantization fp32 -> bf16
  cvt_bf16<<<(int)(WQ / 1024), 256, 0, stream>>>(w_q, wq_b, (int)WQ);
  cvt_bf16<<<(int)(WK / 1024), 256, 0, stream>>>(w_k, wk_b, (int)WK);
  cvt_bf16<<<(int)(WK / 1024), 256, 0, stream>>>(w_v, wv_b, (int)WK);
  cvt_bf16<<<(int)(WQ / 1024), 256, 0, stream>>>(w_o, wo_b, (int)WQ);
  cvt_bf16<<<(int)(WG / 1024), 256, 0, stream>>>(w_g, wg_b, (int)WG);
  cvt_bf16<<<(int)(WG / 1024), 256, 0, stream>>>(w_u, wu_b, (int)WG);
  cvt_bf16<<<(int)(WG / 1024), 256, 0, stream>>>(w_d, wd_b, (int)WG);

  rmsnorm_k<<<SEQ, 256, 0, stream>>>(hidden, ln1, xn);
  gemm_bf16<true><<<dim3(HIDN / 128, SEQ / 128), 256, 0, stream>>>(
      xn, wq_b, nullptr, nullptr, qb, SEQ, HIDN, HIDN);
  gemm_bf16<true><<<dim3((NKV * HD) / 128, SEQ / 128), 256, 0, stream>>>(
      xn, wk_b, nullptr, nullptr, kb, SEQ, NKV * HD, HIDN);
  gemm_bf16<true><<<dim3((NKV * HD) / 128, SEQ / 128), 256, 0, stream>>>(
      xn, wv_b, nullptr, nullptr, vb, SEQ, NKV * HD, HIDN);
  rope_kv<<<SEQ, 64, 0, stream>>>(qb, kb, vb, cosb, sinb, out_k, out_v);
  flash_attn<<<dim3(SEQ / 64, NH), 128, 0, stream>>>(qb, kb, vb, attn);
  gemm_bf16<false><<<dim3(HIDN / 128, SEQ / 128), 256, 0, stream>>>(
      attn, wo_b, hidden, x2, nullptr, SEQ, HIDN, HIDN);
  rmsnorm_k<<<SEQ, 256, 0, stream>>>(x2, ln2, xn);
  gemm_bf16<true><<<dim3(INTER / 128, SEQ / 128), 256, 0, stream>>>(
      xn, wg_b, nullptr, nullptr, gate, SEQ, INTER, HIDN);
  gemm_bf16<true><<<dim3(INTER / 128, SEQ / 128), 256, 0, stream>>>(
      xn, wu_b, nullptr, nullptr, up, SEQ, INTER, HIDN);
  silu_mul<<<(int)(SI / 256), 256, 0, stream>>>(gate, up, hs, (int)SI);
  gemm_bf16<false><<<dim3(HIDN / 128, SEQ / 128), 256, 0, stream>>>(
      hs, wd_b, x2, out_x, nullptr, SEQ, HIDN, INTER);
}
